// ScaledDotProductAttention_74010876444950
// MI455X (gfx1250) — matrix operand order repaired and checked by reading, not compiled, not run on hardware
//
#include <hip/hip_runtime.h>
#include <hip/hip_bf16.h>

typedef _Float16 v16h __attribute__((ext_vector_type(16)));
typedef _Float16 v8h  __attribute__((ext_vector_type(8)));
typedef _Float16 v4h  __attribute__((ext_vector_type(4)));
typedef float    v8f  __attribute__((ext_vector_type(8)));
typedef float    v4f  __attribute__((ext_vector_type(4)));

#define S_LEN   2048
#define DK      128
#define BLOCK_M 128          // 8 waves * 16 query rows
#define TILE_N  32           // keys per iteration
#define KSTR    136          // LDS stride (halfs) for K tile: 272B rows -> conflict-free 16B reads
#define VTSTR   40           // LDS stride (halfs) for V^T: 80B rows -> conflict-free 16B reads
#define FMAXV   3.402823466e38f

__global__ __launch_bounds__(256)
void fa_fwd_kernel(const float* __restrict__ q, const float* __restrict__ k,
                   const float* __restrict__ v, const unsigned char* __restrict__ mask,
                   float* __restrict__ out)
{
    __shared__ _Float16 Ks[32 * KSTR];    // K tile, f16 row-major [key][d]
    __shared__ _Float16 Vt[DK * VTSTR];   // V tile, f16 transposed [d][key]

    const int tid  = threadIdx.x;
    const int wave = tid >> 5;
    const int lane = tid & 31;
    const int lh   = lane >> 4;           // lane half (0/1)
    const int ln   = lane & 15;
    const int bIdx = blockIdx.y;
    const int qrow0 = blockIdx.x * BLOCK_M + wave * 16;

    const float scale = 0.08838834764831845f;   // 1/sqrt(128)

    // ---- Q fragments: B-operand layout for S^T = K * Q^T (B[k=d][n=q] = Q[q,d]) ----
    // lane = query column; lanes 0-15 hold d chunk +0..7,+16..23, lanes 16-31 hold +8..15,+24..31
    v16h qf[4];
    {
        const float* qp = q + ((size_t)bIdx * S_LEN + (size_t)(qrow0 + ln)) * DK;
#pragma unroll
        for (int c = 0; c < 4; ++c) {
            const float* p = qp + c * 32 + lh * 8;
#pragma unroll
            for (int e = 0; e < 16; ++e) qf[c][e] = (_Float16)p[(e & 7) + (e >> 3) * 16];
        }
    }

    v8f o[8];
#pragma unroll
    for (int c = 0; c < 8; ++c) o[c] = (v8f){0.f,0.f,0.f,0.f,0.f,0.f,0.f,0.f};
    float m_run = -FMAXV;
    float l_run = 0.0f;

    const size_t kv_base = (size_t)bIdx * S_LEN * DK;
    const unsigned long long* mrow = (const unsigned long long*)
        (mask + (size_t)bIdx * S_LEN * S_LEN + (size_t)(qrow0 + ln) * S_LEN);

    for (int kb = 0; kb < S_LEN; kb += TILE_N) {
        __syncthreads();
        // ---- cooperative stage of K (row-major f16) and V (transposed f16) ----
#pragma unroll
        for (int i = 0; i < 4; ++i) {
            int idx = tid + 256 * i;          // 1024 float4s = 32 rows * 128 floats
            int row = idx >> 5;               // 0..31 key within tile
            int d4  = (idx & 31) * 4;
            v4f kk = *(const v4f*)(k + kv_base + (size_t)(kb + row) * DK + d4);
            v4h kh; kh[0]=(_Float16)kk[0]; kh[1]=(_Float16)kk[1];
                    kh[2]=(_Float16)kk[2]; kh[3]=(_Float16)kk[3];
            *(v4h*)&Ks[row * KSTR + d4] = kh;
            v4f vv = *(const v4f*)(v + kv_base + (size_t)(kb + row) * DK + d4);
#pragma unroll
            for (int j = 0; j < 4; ++j) Vt[(d4 + j) * VTSTR + row] = (_Float16)vv[j];
        }
        if (kb + TILE_N < S_LEN) {            // L2 prefetch of next tile
            __builtin_prefetch(k + kv_base + (size_t)(kb + TILE_N + (tid >> 3)) * DK + (tid & 7) * 16, 0, 1);
            __builtin_prefetch(v + kv_base + (size_t)(kb + TILE_N + (tid >> 3)) * DK + (tid & 7) * 16, 0, 1);
        }
        __syncthreads();

        // ---- S^T = K * Q^T : two 16-key subtiles, K-dim = 128 in 4 chunks of 32 ----
        v8f s0 = (v8f){0.f,0.f,0.f,0.f,0.f,0.f,0.f,0.f};
        v8f s1 = (v8f){0.f,0.f,0.f,0.f,0.f,0.f,0.f,0.f};
#pragma unroll
        for (int c = 0; c < 4; ++c) {         // subtile 0: keys kb+0..15
            const _Float16* kp = &Ks[ln * KSTR + c * 32 + lh * 8];
            v8h lo = *(const v8h*)kp;
            v8h hi = *(const v8h*)(kp + 16);
            v16h ka;
#pragma unroll
            for (int e = 0; e < 8; ++e) { ka[e] = lo[e]; ka[8 + e] = hi[e]; }
            s0 = __builtin_amdgcn_wmma_f32_16x16x32_f16(false, ka, false, qf[c],
                                                        (short)0, s0, false, false);
        }
#pragma unroll
        for (int c = 0; c < 4; ++c) {         // subtile 1: keys kb+16..31
            const _Float16* kp = &Ks[(16 + ln) * KSTR + c * 32 + lh * 8];
            v8h lo = *(const v8h*)kp;
            v8h hi = *(const v8h*)(kp + 16);
            v16h ka;
#pragma unroll
            for (int e = 0; e < 8; ++e) { ka[e] = lo[e]; ka[8 + e] = hi[e]; }
            s1 = __builtin_amdgcn_wmma_f32_16x16x32_f16(false, ka, false, qf[c],
                                                        (short)0, s1, false, false);
        }

        // ---- mask + online softmax (lane = query; keys live in VGPR dim + lane half) ----
        unsigned long long mb0 = mrow[(kb >> 3) + lh];       // bytes kb + lh*8 .. +7
        unsigned long long mb1 = mrow[(kb >> 3) + lh + 2];   // bytes kb+16 + lh*8 .. +7
        float p0[8], p1[8];
        float tmax = -FMAXV;
#pragma unroll
        for (int r = 0; r < 8; ++r) {
            float a = s0[r] * scale; if ((mb0 >> (8 * r)) & 0xffull) a = -FMAXV;
            float b = s1[r] * scale; if ((mb1 >> (8 * r)) & 0xffull) b = -FMAXV;
            p0[r] = a; p1[r] = b;
            tmax = fmaxf(tmax, fmaxf(a, b));
        }
        tmax = fmaxf(tmax, __shfl_xor(tmax, 16, 32));        // combine lane halves
        float m_new = fmaxf(m_run, tmax);
        float alpha = __expf(m_run - m_new);
        float psum = 0.f;
        v16h pa;                                             // P as A-fragment, direct
#pragma unroll
        for (int r = 0; r < 8; ++r) {
            float e0 = __expf(p0[r] - m_new);
            float e1 = __expf(p1[r] - m_new);
            psum += e0 + e1;
            pa[r]     = (_Float16)e0;
            pa[8 + r] = (_Float16)e1;
        }
        psum += __shfl_xor(psum, 16, 32);
        l_run = l_run * alpha + psum;
        m_run = m_new;

        // broadcast alpha from lane space (lane=query) into VGPR-row space of O
        float arow[8];
#pragma unroll
        for (int r = 0; r < 8; ++r) arow[r] = __shfl(alpha, r + lh * 8, 32);

        // ---- O += P * V over 8 d-column chunks ----
#pragma unroll
        for (int c2 = 0; c2 < 8; ++c2) {
#pragma unroll
            for (int r = 0; r < 8; ++r) o[c2][r] *= arow[r];
            const _Float16* vp = &Vt[(c2 * 16 + ln) * VTSTR + lh * 8];
            v8h lo = *(const v8h*)vp;
            v8h hi = *(const v8h*)(vp + 16);
            v16h vb;
#pragma unroll
            for (int e = 0; e < 8; ++e) { vb[e] = lo[e]; vb[8 + e] = hi[e]; }
            o[c2] = __builtin_amdgcn_wmma_f32_16x16x32_f16(false, pa, false, vb,
                                                           (short)0, o[c2], false, false);
        }
    }

    // ---- normalize and write out ----
    float linv[8];
#pragma unroll
    for (int r = 0; r < 8; ++r) {
        float lr = __shfl(l_run, r + lh * 8, 32);
        linv[r] = 1.0f / lr;
    }
    float* op = out + ((size_t)bIdx * S_LEN + (size_t)qrow0) * DK;
#pragma unroll
    for (int c2 = 0; c2 < 8; ++c2) {
#pragma unroll
        for (int r = 0; r < 8; ++r) {
            op[(size_t)(r + lh * 8) * DK + c2 * 16 + ln] = o[c2][r] * linv[r];
        }
    }
}

extern "C" void kernel_launch(void* const* d_in, const int* in_sizes, int n_in,
                              void* d_out, int out_size, void* d_ws, size_t ws_size,
                              hipStream_t stream) {
    const float* q = (const float*)d_in[0];
    const float* k = (const float*)d_in[1];
    const float* v = (const float*)d_in[2];
    const unsigned char* mask = (const unsigned char*)d_in[3];  // jnp.bool_ -> 1 byte
    float* out = (float*)d_out;

    dim3 grid(S_LEN / BLOCK_M, 16 /*B*/, 1);
    dim3 block(256, 1, 1);
    fa_fwd_kernel<<<grid, block, 0, stream>>>(q, k, v, mask, out);
}
